// VQEmbedding_55911884259971
// MI455X (gfx1250) — matrix operand order repaired and checked by reading, not compiled, not run on hardware
//
#include <hip/hip_runtime.h>

#define N_ROWS 32768
#define D_DIM  512
#define K_CB   8192
#define BM     128
#define BN     128
#define BKS    64      // k-elements staged per LDS buffer (2 WMMA k-substeps)

typedef __attribute__((ext_vector_type(16))) __bf16 v16bf;
typedef __attribute__((ext_vector_type(8)))  __bf16 v8bf;
typedef __attribute__((ext_vector_type(8)))  float  v8f;
typedef __attribute__((ext_vector_type(4)))  int    v4i;

#if defined(__HIP_DEVICE_COMPILE__) &&                                   \
    __has_builtin(__builtin_amdgcn_global_load_async_to_lds_b128) &&     \
    __has_builtin(__builtin_amdgcn_s_wait_asynccnt)
#define USE_ASYNC_LDS 1
// pointer-to-AS(1)-v4i and pointer-to-AS(3)-v4i, per the builtin prototype
// "vV4i*1V4i*3IiIi" (src_global, dst_lds, imm offset, imm cpol)
typedef __attribute__((address_space(1))) v4i* gptr_v4i;
typedef __attribute__((address_space(3))) v4i* lptr_v4i;
#else
#define USE_ASYNC_LDS 0
#endif

// ---------------------------------------------------------------------------
// Kernel 1: convert z (fp32 -> bf16), vectorized 4 elements/thread
// ---------------------------------------------------------------------------
__global__ __launch_bounds__(256) void cvt_z_kernel(const float* __restrict__ z,
                                                    __bf16* __restrict__ zb) {
  const size_t i = (size_t)blockIdx.x * 1024 + (size_t)threadIdx.x * 4;
  float4 v = *(const float4*)(z + i);
  zb[i + 0] = (__bf16)v.x;
  zb[i + 1] = (__bf16)v.y;
  zb[i + 2] = (__bf16)v.z;
  zb[i + 3] = (__bf16)v.w;
}

// ---------------------------------------------------------------------------
// Kernel 2: convert codebook (fp32 -> bf16) + exact fp32 ||c||^2 per row
// ---------------------------------------------------------------------------
__global__ __launch_bounds__(256) void cvt_cb_kernel(const float* __restrict__ cb,
                                                     __bf16* __restrict__ cbb,
                                                     float* __restrict__ cbsqr) {
  __shared__ float red[256];
  const int row = blockIdx.x;
  const int t   = threadIdx.x;
  const float* r = cb + (size_t)row * D_DIM;
  float s = 0.f;
#pragma unroll
  for (int j = 0; j < D_DIM / 256; ++j) {
    const int i = t + j * 256;
    const float v = r[i];
    cbb[(size_t)row * D_DIM + i] = (__bf16)v;
    s += v * v;
  }
  red[t] = s;
  __syncthreads();
  for (int off = 128; off > 0; off >>= 1) {
    if (t < off) red[t] += red[t + off];
    __syncthreads();
  }
  if (t == 0) cbsqr[row] = red[0];
}

// ---------------------------------------------------------------------------
// Kernel 3: fused bf16-WMMA GEMM + argmin. Double-buffered async-to-LDS B
// staging, pipelined ACROSS n-tile boundaries (the only cold stage is the
// very first chunk). Block = 256 threads = 8 waves; block owns BM=128 z-rows;
// wave w owns rows [w*16, w*16+16). 64 codebook tiles of BN=128; K=512
// accumulated via v_wmma_f32_16x16x32_bf16; rank-dist = ||c||^2 - 2*(z.c).
// ---------------------------------------------------------------------------
__global__ __launch_bounds__(256) void vq_argmin_kernel(
    const __bf16* __restrict__ zb, const __bf16* __restrict__ cbb,
    const float* __restrict__ cbsqr, int* __restrict__ idx_out) {
  // two 16 KB B buffers: 128 rows x 64 bf16 (128 B/row)
  __shared__ __align__(64) __bf16 Bs[2][BN * BKS];

  const int tid    = threadIdx.x;
  const int wave   = tid >> 5;
  const int lane   = tid & 31;
  const int half   = lane >> 4;   // 0: lanes 0-15, 1: lanes 16-31
  const int l16    = lane & 15;
  const int m_base = blockIdx.x * BM + wave * 16;

  // A-fragment source (contiguous k order used here, not the ISA table order):
  //   lanes 0-15 : row M=l16, K = [0..15]  (contiguous, matches B)
  //   lanes 16-31: row M=l16, K = [16..31]
  const __bf16* arow = zb + (size_t)(m_base + l16) * D_DIM + half * 16;

  // staging assignment: 1024 x 16B pieces per 16 KB buffer; each of the 256
  // threads deposits 4 pieces (elem offsets +0, +8, +32, +40) in one row.
  const int srow  = tid >> 1;        // 0..127  (2 threads per row)
  const int scol0 = (tid & 1) * 16;  // elem offset within row: 0 or 16

  float minv[8];
  int   mini[8];
#pragma unroll
  for (int r = 0; r < 8; ++r) { minv[r] = 3.4e38f; mini[r] = 0; }

  const int NSTAGE = D_DIM / BKS;   // 8 k-chunks per n-tile (even!)
  const int NT     = K_CB / BN;     // 64 n-tiles

  // ---- stage helper: deposit 4x16B per thread straight into LDS ----------
#if USE_ASYNC_LDS
#define STAGE_B(buf, nbase, kc)                                                \
  {                                                                            \
    const __bf16* g = cbb + (size_t)((nbase) + srow) * D_DIM + (kc)*BKS + scol0; \
    __bf16* l = &Bs[(buf)][srow * BKS + scol0];                                \
    __builtin_amdgcn_global_load_async_to_lds_b128(                            \
        (gptr_v4i)(void*)g, (lptr_v4i)(void*)l, 0, 0);                         \
    __builtin_amdgcn_global_load_async_to_lds_b128(                            \
        (gptr_v4i)(void*)(g + 8), (lptr_v4i)(void*)(l + 8), 0, 0);             \
    __builtin_amdgcn_global_load_async_to_lds_b128(                            \
        (gptr_v4i)(void*)(g + 32), (lptr_v4i)(void*)(l + 32), 0, 0);           \
    __builtin_amdgcn_global_load_async_to_lds_b128(                            \
        (gptr_v4i)(void*)(g + 40), (lptr_v4i)(void*)(l + 40), 0, 0);           \
  }
#define STAGE_WAIT() __builtin_amdgcn_s_wait_asynccnt(0)
#else
#define STAGE_B(buf, nbase, kc)                                                \
  {                                                                            \
    const __bf16* g = cbb + (size_t)((nbase) + srow) * D_DIM + (kc)*BKS + scol0; \
    __bf16* l = &Bs[(buf)][srow * BKS + scol0];                                \
    *(int4*)l        = *(const int4*)g;                                        \
    *(int4*)(l + 8)  = *(const int4*)(g + 8);                                  \
    *(int4*)(l + 32) = *(const int4*)(g + 32);                                 \
    *(int4*)(l + 40) = *(const int4*)(g + 40);                                 \
  }
#define STAGE_WAIT() ((void)0)
#endif

  // cold prologue: fill buffer 0 with (tile 0, chunk 0) — the only unhidden
  // stage in the whole kernel.
  STAGE_B(0, 0, 0);
  STAGE_WAIT();
  __syncthreads();

  for (int nt = 0; nt < NT; ++nt) {
    const int n_base = nt * BN;

    // speculative prefetch of the next tile into GL2
    if (nt + 1 < NT)
      __builtin_prefetch(cbb + (size_t)(n_base + BN + (tid & (BN - 1))) * D_DIM, 0, 1);

    v8f C[8] = {};  // columns [n_base + j*16 + l16]

    for (int kc = 0; kc < NSTAGE; ++kc) {
      const int cur = kc & 1;
      // stage the next chunk — crossing into the next n-tile at kc==7.
      // NSTAGE even => buffer parity stays consistent across tiles.
      if (kc + 1 < NSTAGE)      STAGE_B(cur ^ 1, n_base, kc + 1)
      else if (nt + 1 < NT)     STAGE_B(cur ^ 1, n_base + BN, 0)

      // two 32-wide WMMA k-substeps on the current buffer
#pragma unroll
      for (int s = 0; s < 2; ++s) {
        const int k0 = kc * BKS + s * 32;

        const v8bf alo = *(const v8bf*)(arow + k0);
        const v8bf ahi = *(const v8bf*)(arow + k0 + 8);
        v16bf afrag;
#pragma unroll
        for (int e = 0; e < 8; ++e) { afrag[e] = alo[e]; afrag[e + 8] = ahi[e]; }

        // B frag (contiguous k order, same as A): lane = column n = l16,
        // K=[0..15] (lanes 0-15) / K=[16..31] (lanes 16-31), contiguous.
#pragma unroll
        for (int j = 0; j < 8; ++j) {
          const v16bf bfrag =
              *(const v16bf*)&Bs[cur][(j * 16 + l16) * BKS + s * 32 + half * 16];
          C[j] = __builtin_amdgcn_wmma_f32_16x16x32_bf16(
              false, afrag, false, bfrag, (short)0, C[j], false, false);
        }
      }

      STAGE_WAIT();      // our async deposits (next chunk) are done
      __syncthreads();   // everyone's deposits visible; buf cur free to reuse
    }

    // Epilogue: rank-dist = ||c||^2 - 2*dot ; running argmin.
    // C layout: slot r, lanes 0-15 -> row r ; lanes 16-31 -> row r+8.
#pragma unroll
    for (int j = 0; j < 8; ++j) {
      const int   n   = n_base + j * 16 + l16;
      const float nsq = cbsqr[n];
#pragma unroll
      for (int r = 0; r < 8; ++r) {
        const float d = __builtin_fmaf(-2.f, C[j][r], nsq);
        if (d < minv[r]) { minv[r] = d; mini[r] = n; }
      }
    }
  }
#undef STAGE_B
#undef STAGE_WAIT

  // Reduce (min, argmin) across each 16-lane half; first-index tie-break.
#pragma unroll
  for (int r = 0; r < 8; ++r) {
    float v = minv[r];
    int   i = mini[r];
#pragma unroll
    for (int m = 1; m < 16; m <<= 1) {
      const float ov = __shfl_xor(v, m, 32);
      const int   oi = __shfl_xor(i, m, 32);
      if (ov < v || (ov == v && oi < i)) { v = ov; i = oi; }
    }
    if (l16 == 0) idx_out[m_base + r + half * 8] = i;
  }
}

// ---------------------------------------------------------------------------
// Kernel 4: exact fp32 loss: out[n] = 1.25 * sum((z[n] - cb[idx[n]])^2)
// ---------------------------------------------------------------------------
__global__ __launch_bounds__(256) void vq_loss_kernel(
    const float* __restrict__ z, const float* __restrict__ cb,
    const int* __restrict__ idx, float* __restrict__ out) {
  const int wave = threadIdx.x >> 5;
  const int lane = threadIdx.x & 31;
  const int row  = blockIdx.x * 8 + wave;
  const float* zr = z  + (size_t)row * D_DIM;
  const float* cr = cb + (size_t)idx[row] * D_DIM;
  float s = 0.f;
#pragma unroll
  for (int i = lane * 4; i < D_DIM; i += 128) {
    const float4 a = *(const float4*)(zr + i);
    const float4 b = *(const float4*)(cr + i);
    const float dx = a.x - b.x, dy = a.y - b.y, dz = a.z - b.z, dw = a.w - b.w;
    s += dx * dx + dy * dy + dz * dz + dw * dw;
  }
#pragma unroll
  for (int m = 16; m >= 1; m >>= 1) s += __shfl_xor(s, m, 32);
  if (lane == 0) out[row] = 1.25f * s;  // vq_loss + 0.25 * commitment_loss
}

// ---------------------------------------------------------------------------
// Launch
// ---------------------------------------------------------------------------
extern "C" void kernel_launch(void* const* d_in, const int* in_sizes, int n_in,
                              void* d_out, int out_size, void* d_ws, size_t ws_size,
                              hipStream_t stream) {
  const float* z  = (const float*)d_in[0];   // [N, D] fp32
  const float* cb = (const float*)d_in[1];   // [K, D] fp32
  float* out = (float*)d_out;                // [N] fp32

  // workspace layout (~40.2 MB):
  //   zb   : N*D   bf16  (32 MB)
  //   cbb  : K*D   bf16  ( 8 MB)
  //   cbsq : K     fp32  (32 KB)
  //   idx  : N     int32 (128 KB)
  char* ws = (char*)d_ws;
  __bf16* zb   = (__bf16*)ws;
  __bf16* cbb  = (__bf16*)(ws + (size_t)N_ROWS * D_DIM * 2);
  float*  cbsq = (float*)(ws + (size_t)N_ROWS * D_DIM * 2 + (size_t)K_CB * D_DIM * 2);
  int*    idx  = (int*)((char*)cbsq + (size_t)K_CB * 4);

  cvt_z_kernel<<<(N_ROWS * D_DIM) / 1024, 256, 0, stream>>>(z, zb);
  cvt_cb_kernel<<<K_CB, 256, 0, stream>>>(cb, cbb, cbsq);
  vq_argmin_kernel<<<N_ROWS / BM, 256, 0, stream>>>(zb, cbb, cbsq, idx);
  vq_loss_kernel<<<N_ROWS / 8, 256, 0, stream>>>(z, cb, idx, out);
}
